// FGTMultiHeadAttention_15857019257343
// MI455X (gfx1250) — matrix operand order repaired and checked by reading, not compiled, not run on hardware
//
#include <hip/hip_runtime.h>

// ---------------------------------------------------------------------------
// MHA (B=2, T=2048, D=1024, H=16, hd=64) on gfx1250 with v_wmma_f32_16x16x32_f16
// Pipeline: cvt(fp32->fp16) -> qkv GEMM+RoPE -> flash attention -> o-proj.
// All GEMM loops are ping-pong double-buffered (x2 unroll, no rotation copies).
// ---------------------------------------------------------------------------

typedef __attribute__((ext_vector_type(16))) _Float16 v16h;
typedef __attribute__((ext_vector_type(8)))  _Float16 v8h;
typedef __attribute__((ext_vector_type(4)))  _Float16 v4h;
typedef __attribute__((ext_vector_type(8)))  float    v8f;

#define DEV __device__ __forceinline__

static constexpr int Bsz = 2;
static constexpr int T   = 2048;
static constexpr int D   = 1024;
static constexpr int H   = 16;
static constexpr int HD  = 64;

DEV v8f wmma32(v16h a, v16h b, v8f c) {
  return __builtin_amdgcn_wmma_f32_16x16x32_f16(
      /*neg_a=*/false, a, /*neg_b=*/false, b,
      /*c_mod=*/(short)0, c, /*reuse_a=*/false, /*reuse_b=*/false);
}

// A-operand (16x32, M-major): lane = M row; lanes 0-15 hold K {0-7,16-23},
// lanes 16-31 hold K {8-15,24-31}.
DEV v16h load_tileA(const _Float16* base, int ld) {
  int lane = threadIdx.x & 31;
  const _Float16* p = base + (size_t)(lane & 15) * ld + (lane >> 4) * 8;
  union { v16h v; v8h h[2]; } u;
  u.h[0] = *(const v8h*)(p);
  u.h[1] = *(const v8h*)(p + 16);
  return u.v;
}

// B-operand (32x16 = K x N) loaded from a row-major [N x K] view (= B^T):
// lane = N row; lanes 0-15 hold K {0-7,16-23}, lanes 16-31 hold K {8-15,24-31}.
DEV v16h load_tileB(const _Float16* base, int ld) {
  int lane = threadIdx.x & 31;
  const _Float16* p = base + (size_t)(lane & 15) * ld + (lane >> 4) * 8;
  union { v16h v; v8h h[2]; } u;
  u.h[0] = *(const v8h*)(p);
  u.h[1] = *(const v8h*)(p + 16);
  return u.v;
}

// ---------------------------------------------------------------------------
// Kernel 0: fp32 -> fp16 conversion (4 elements / thread)
// ---------------------------------------------------------------------------
__global__ __launch_bounds__(256) void cvt_kernel(const float* __restrict__ src,
                                                  _Float16* __restrict__ dst,
                                                  int n) {
  int i = (blockIdx.x * 256 + threadIdx.x) * 4;
  if (i + 3 < n) {
    float4 f = *(const float4*)(src + i);
    v4h h;
    h[0] = (_Float16)f.x; h[1] = (_Float16)f.y;
    h[2] = (_Float16)f.z; h[3] = (_Float16)f.w;
    *(v4h*)(dst + i) = h;
  }
}

// ---------------------------------------------------------------------------
// Shared GEMM stage: 32(M) x 64(N) per wave over K=D, ping-pong buffered.
// pA0/pA1 = row-major A rows m0 / m0+16; pB[j] = row-major B^T rows n0+16j.
// ---------------------------------------------------------------------------
struct GemmBufs { v16h a0, a1, b[4]; };

DEV void gemm_load(GemmBufs& g, const _Float16* pA0, const _Float16* pA1,
                   const _Float16* const* pB, int k) {
  g.a0 = load_tileA(pA0 + k, D);
  g.a1 = load_tileA(pA1 + k, D);
#pragma unroll
  for (int j = 0; j < 4; ++j) g.b[j] = load_tileB(pB[j] + k, D);
}

DEV void gemm_mma(const GemmBufs& g, v8f acc[2][4]) {
#pragma unroll
  for (int j = 0; j < 4; ++j) acc[0][j] = wmma32(g.a0, g.b[j], acc[0][j]);
#pragma unroll
  for (int j = 0; j < 4; ++j) acc[1][j] = wmma32(g.a1, g.b[j], acc[1][j]);
}

// ---------------------------------------------------------------------------
// Kernel 1: QKV projection + RoPE.  grid = (M/32, D/512, 3), block = 256.
//   which: 0=q (rope), 1=k (rope), 2=v (stored transposed [b,h,d,t])
// ---------------------------------------------------------------------------
__global__ __launch_bounds__(256) void qkv_kernel(
    const _Float16* __restrict__ x16,
    const _Float16* __restrict__ wq16, const _Float16* __restrict__ wk16,
    const _Float16* __restrict__ wv16,
    _Float16* __restrict__ Q16, _Float16* __restrict__ K16,
    _Float16* __restrict__ Vt16) {
  int wid   = threadIdx.x >> 5;
  int lane  = threadIdx.x & 31;
  int lh    = lane >> 4;
  int m0    = blockIdx.x * 32;
  int n0    = blockIdx.y * 512 + wid * 64;
  int which = blockIdx.z;
  const _Float16* W = (which == 0) ? wq16 : (which == 1) ? wk16 : wv16;

  const _Float16* pA0 = x16 + (size_t)m0 * D;
  const _Float16* pA1 = x16 + (size_t)(m0 + 16) * D;
  const _Float16* pB[4];
#pragma unroll
  for (int j = 0; j < 4; ++j) pB[j] = W + (size_t)(n0 + 16 * j) * D;

  v8f acc[2][4];
#pragma unroll
  for (int mi = 0; mi < 2; ++mi)
#pragma unroll
    for (int j = 0; j < 4; ++j) acc[mi][j] = v8f{};

  GemmBufs g0, g1;
  gemm_load(g0, pA0, pA1, pB, 0);
  for (int k0 = 0; k0 < D; k0 += 64) {
    gemm_load(g1, pA0, pA1, pB, k0 + 32);     // prefetch while computing g0
    gemm_mma(g0, acc);
    gemm_load(g0, pA0, pA1, pB, (k0 + 64) & (D - 1));  // wrap on last (benign)
    gemm_mma(g1, acc);
  }

  int h = (n0 >> 6) & (H - 1);
#pragma unroll
  for (int mi = 0; mi < 2; ++mi) {
    int mb = m0 + 16 * mi;
    if (which < 2) {  // RoPE: pair (d, d+32) -> acc[mi][j] with acc[mi][j+2]
#pragma unroll
      for (int j = 0; j < 2; ++j) {
        float p   = (float)(16 * j + (lane & 15));
        float inv = __expf(-0.28782313662425572f * p);  // 10000^(-p/32)
#pragma unroll
        for (int e = 0; e < 8; ++e) {
          int   m   = mb + e + 8 * lh;
          float ang = (float)(m & (T - 1)) * inv;
          float c = __cosf(ang), s = __sinf(ang);
          float x1 = acc[mi][j][e], x2 = acc[mi][j + 2][e];
          acc[mi][j][e]     = x1 * c - x2 * s;
          acc[mi][j + 2][e] = x2 * c + x1 * s;
        }
      }
    }
#pragma unroll
    for (int j = 0; j < 4; ++j) {
#pragma unroll
      for (int e = 0; e < 8; ++e) {
        int m = mb + e + 8 * lh;
        int t = m & (T - 1);
        int b = m >> 11;
        int d = 16 * j + (lane & 15);
        _Float16 val = (_Float16)acc[mi][j][e];
        if (which == 2)
          Vt16[((size_t)(b * H + h) * HD + d) * T + t] = val;
        else if (which == 0)
          Q16[((size_t)(b * H + h) * T + t) * HD + d] = val;
        else
          K16[((size_t)(b * H + h) * T + t) * HD + d] = val;
      }
    }
  }
}

// ---------------------------------------------------------------------------
// Kernel 2: flash attention.  grid = B*H*(T/16)/8, block = 256 (8 waves),
// one wave per 16-query tile; keys streamed in ping-pong chunks of 32.
// ---------------------------------------------------------------------------
#define NW 8

struct AttnChunk { v16h k[4]; v16h v[4]; };

DEV void chunk_load(AttnChunk& c, const _Float16* kbase, const _Float16* vbase,
                    int kj) {
  c.k[0] = load_tileB(kbase + (size_t)kj * HD, HD);
  c.k[1] = load_tileB(kbase + (size_t)kj * HD + 32, HD);
  c.k[2] = load_tileB(kbase + (size_t)(kj + 16) * HD, HD);
  c.k[3] = load_tileB(kbase + (size_t)(kj + 16) * HD + 32, HD);
#pragma unroll
  for (int j = 0; j < 4; ++j)
    c.v[j] = load_tileB(vbase + (size_t)(16 * j) * T + kj, T);
}

DEV void chunk_process(const AttnChunk& c, v16h A0, v16h A1, v8f accO[4],
                       float mrow[8], float lrow[8], _Float16* myLds) {
  int lane = threadIdx.x & 31;
  int lh   = lane >> 4;

  v8f S0 = v8f{}, S1 = v8f{};
  S0 = wmma32(A0, c.k[0], S0);
  S0 = wmma32(A1, c.k[1], S0);
  S1 = wmma32(A0, c.k[2], S1);
  S1 = wmma32(A1, c.k[3], S1);

  float scale[8];
#pragma unroll
  for (int e = 0; e < 8; ++e) {
    float v = fmaxf(S0[e], S1[e]);
#pragma unroll
    for (int off = 1; off < 16; off <<= 1) v = fmaxf(v, __shfl_xor(v, off));
    float mn = fmaxf(mrow[e], v);
    scale[e] = __expf(mrow[e] - mn);
    mrow[e]  = mn;
    float p0 = __expf(S0[e] - mn);
    float p1 = __expf(S1[e] - mn);
    S0[e] = p0; S1[e] = p1;
    float r = p0 + p1;
#pragma unroll
    for (int off = 1; off < 16; off <<= 1) r += __shfl_xor(r, off);
    lrow[e] = lrow[e] * scale[e] + r;
  }
#pragma unroll
  for (int j = 0; j < 4; ++j)
#pragma unroll
    for (int e = 0; e < 8; ++e) accO[j][e] = accO[j][e] * scale[e];

  // P (C-layout, lane = column) -> LDS -> reload in A-operand layout.
  int col = lane & 15;
#pragma unroll
  for (int e = 0; e < 8; ++e) {
    int row = e + 8 * lh;
    myLds[row * 32 + col]      = (_Float16)S0[e];
    myLds[row * 32 + 16 + col] = (_Float16)S1[e];
  }
  asm volatile("s_wait_dscnt 0" ::: "memory");
  v16h Ap = load_tileA(myLds, 32);

#pragma unroll
  for (int j = 0; j < 4; ++j) accO[j] = wmma32(Ap, c.v[j], accO[j]);
}

__global__ __launch_bounds__(256) void attn_kernel(
    const _Float16* __restrict__ Q16, const _Float16* __restrict__ K16,
    const _Float16* __restrict__ Vt16, _Float16* __restrict__ O16) {
  __shared__ _Float16 smem[NW][16 * 32];  // per-wave P tile (16 x 32 f16)

  int wid  = threadIdx.x >> 5;
  int lane = threadIdx.x & 31;
  int lh   = lane >> 4;
  int tile = blockIdx.x * NW + wid;
  int qt   = tile & (T / 16 - 1);
  int bh   = tile >> 7;  // T/16 == 128
  const _Float16* qbase = Q16 + ((size_t)bh * T + qt * 16) * HD;
  const _Float16* kbase = K16 + (size_t)bh * T * HD;
  const _Float16* vbase = Vt16 + (size_t)bh * HD * T;
  _Float16* myLds = smem[wid];

  v16h A0 = load_tileA(qbase, HD);
  v16h A1 = load_tileA(qbase + 32, HD);
#pragma unroll
  for (int i = 0; i < 16; ++i) {  // fold 1/sqrt(hd) into Q
    A0[i] = A0[i] * (_Float16)0.125f;
    A1[i] = A1[i] * (_Float16)0.125f;
  }

  float mrow[8], lrow[8];
  v8f accO[4] = {v8f{}, v8f{}, v8f{}, v8f{}};
#pragma unroll
  for (int e = 0; e < 8; ++e) { mrow[e] = -1e30f; lrow[e] = 0.f; }

  AttnChunk c0, c1;
  chunk_load(c0, kbase, vbase, 0);
  for (int kj = 0; kj < T; kj += 64) {
    if (kj + 64 < T) {  // warm L2 one stage ahead (global_prefetch_b8)
      __builtin_prefetch(kbase + (size_t)(kj + 64) * HD, 0, 1);
      __builtin_prefetch(vbase + (size_t)(kj + 64), 0, 1);
    }
    chunk_load(c1, kbase, vbase, kj + 32);  // in flight during c0 compute
    chunk_process(c0, A0, A1, accO, mrow, lrow, myLds);
    chunk_load(c0, kbase, vbase, (kj + 64) & (T - 1));  // wrap benign
    chunk_process(c1, A0, A1, accO, mrow, lrow, myLds);
  }

  int h = bh & (H - 1);
  int b = bh >> 4;
#pragma unroll
  for (int j = 0; j < 4; ++j) {
#pragma unroll
    for (int e = 0; e < 8; ++e) {
      int t = qt * 16 + e + 8 * lh;
      int d = 16 * j + (lane & 15);
      O16[((size_t)(b * T + t)) * D + h * HD + d] =
          (_Float16)(accO[j][e] * (1.0f / lrow[e]));
    }
  }
}

// ---------------------------------------------------------------------------
// Kernel 3: output projection  out = O @ w_o^T + b_o  (fp32), 32x64 per wave.
// ---------------------------------------------------------------------------
__global__ __launch_bounds__(256) void oproj_kernel(
    const _Float16* __restrict__ A16, const _Float16* __restrict__ Wo16,
    const float* __restrict__ bias, float* __restrict__ out) {
  int wid  = threadIdx.x >> 5;
  int lane = threadIdx.x & 31;
  int lh   = lane >> 4;
  int m0   = blockIdx.x * 32;
  int n0   = blockIdx.y * 512 + wid * 64;

  const _Float16* pA0 = A16 + (size_t)m0 * D;
  const _Float16* pA1 = A16 + (size_t)(m0 + 16) * D;
  const _Float16* pB[4];
#pragma unroll
  for (int j = 0; j < 4; ++j) pB[j] = Wo16 + (size_t)(n0 + 16 * j) * D;

  v8f acc[2][4];
#pragma unroll
  for (int mi = 0; mi < 2; ++mi)
#pragma unroll
    for (int j = 0; j < 4; ++j) acc[mi][j] = v8f{};

  GemmBufs g0, g1;
  gemm_load(g0, pA0, pA1, pB, 0);
  for (int k0 = 0; k0 < D; k0 += 64) {
    gemm_load(g1, pA0, pA1, pB, k0 + 32);
    gemm_mma(g0, acc);
    gemm_load(g0, pA0, pA1, pB, (k0 + 64) & (D - 1));
    gemm_mma(g1, acc);
  }

#pragma unroll
  for (int mi = 0; mi < 2; ++mi) {
#pragma unroll
    for (int j = 0; j < 4; ++j) {
      int   n  = n0 + 16 * j + (lane & 15);
      float bj = bias[n];
#pragma unroll
      for (int e = 0; e < 8; ++e) {
        int m = m0 + 16 * mi + e + 8 * lh;
        out[(size_t)m * D + n] = acc[mi][j][e] + bj;
      }
    }
  }
}

// ---------------------------------------------------------------------------
extern "C" void kernel_launch(void* const* d_in, const int* in_sizes, int n_in,
                              void* d_out, int out_size, void* d_ws,
                              size_t ws_size, hipStream_t stream) {
  const float* x  = (const float*)d_in[0];
  const float* wq = (const float*)d_in[1];
  const float* wk = (const float*)d_in[2];
  const float* wv = (const float*)d_in[3];
  const float* wo = (const float*)d_in[4];
  const float* bo = (const float*)d_in[5];
  float* out = (float*)d_out;

  char* ws = (char*)d_ws;
  const size_t XB = (size_t)Bsz * T * D * sizeof(_Float16);  // 8 MiB
  const size_t WB = (size_t)D * D * sizeof(_Float16);        // 2 MiB
  _Float16* x16  = (_Float16*)(ws);
  _Float16* wq16 = (_Float16*)(ws + XB);
  _Float16* wk16 = (_Float16*)(ws + XB + WB);
  _Float16* wv16 = (_Float16*)(ws + XB + 2 * WB);
  _Float16* wo16 = (_Float16*)(ws + XB + 3 * WB);
  _Float16* Q16  = (_Float16*)(ws + XB + 4 * WB);
  _Float16* K16  = (_Float16*)(ws + 2 * XB + 4 * WB);
  _Float16* Vt16 = (_Float16*)(ws + 3 * XB + 4 * WB);
  _Float16* O16  = (_Float16*)(ws + 4 * XB + 4 * WB);

  const int nx = Bsz * T * D;  // 4 Mi
  const int nw = D * D;        // 1 Mi
  cvt_kernel<<<nx / (256 * 4), 256, 0, stream>>>(x, x16, nx);
  cvt_kernel<<<nw / (256 * 4), 256, 0, stream>>>(wq, wq16, nw);
  cvt_kernel<<<nw / (256 * 4), 256, 0, stream>>>(wk, wk16, nw);
  cvt_kernel<<<nw / (256 * 4), 256, 0, stream>>>(wv, wv16, nw);
  cvt_kernel<<<nw / (256 * 4), 256, 0, stream>>>(wo, wo16, nw);

  qkv_kernel<<<dim3(Bsz * T / 32, D / 512, 3), 256, 0, stream>>>(
      x16, wq16, wk16, wv16, Q16, K16, Vt16);

  attn_kernel<<<Bsz * H * (T / 16) / NW, 256, 0, stream>>>(Q16, K16, Vt16, O16);

  oproj_kernel<<<dim3(Bsz * T / 32, D / 512), 256, 0, stream>>>(O16, wo16, bo,
                                                                out);
}
